// Mask2FormerHungarianMatcher_42614665511136
// MI455X (gfx1250) — matrix operand order repaired and checked by reading, not compiled, not run on hardware
//
#include <hip/hip_runtime.h>
#include <hip/hip_bf16.h>
#include <math.h>

// ---------------- problem constants ----------------
#define B_  2
#define Q_  100
#define T_  30
#define L_  134
#define P_  12544            // 49 * 256 exactly
#define VOL_ (64*64*64)
#define QPAD_ 112            // 7 * 16
#define TPAD_ 32             // 2 * 16
#define CLAMP_ 1e10f

typedef __attribute__((ext_vector_type(16))) _Float16 v16h;
typedef __attribute__((ext_vector_type(8)))  _Float16 v8h;
typedef __attribute__((ext_vector_type(8)))  float    v8f;

// ---------------- trilinear sampler (matches torch grid_sample,
// align_corners=False, padding zeros; x->D, y->W, z->H) ----------------
__device__ __forceinline__ float sample_tri(const float* __restrict__ vol,
                                            float c0, float c1, float c2) {
  // unnormalize: ((2c-1+1)*S - 1)/2 = c*S - 0.5
  float x = c0 * 64.0f - 0.5f;   // indexes D (innermost)
  float y = c1 * 64.0f - 0.5f;   // indexes W
  float z = c2 * 64.0f - 0.5f;   // indexes H (outermost)
  float x0f = floorf(x), y0f = floorf(y), z0f = floorf(z);
  float wx = x - x0f, wy = y - y0f, wz = z - z0f;
  int x0 = (int)x0f, y0 = (int)y0f, z0 = (int)z0f;
  float out = 0.0f;
#pragma unroll
  for (int dz = 0; dz < 2; ++dz) {
#pragma unroll
    for (int dy = 0; dy < 2; ++dy) {
#pragma unroll
      for (int dx = 0; dx < 2; ++dx) {
        int zi = z0 + dz, yi = y0 + dy, xi = x0 + dx;
        bool valid = (zi >= 0) & (zi < 64) & (yi >= 0) & (yi < 64) &
                     (xi >= 0) & (xi < 64);
        int zc = min(max(zi, 0), 63);
        int yc = min(max(yi, 0), 63);
        int xc = min(max(xi, 0), 63);
        float v = vol[((size_t)zc * 64 + yc) * 64 + xc];
        v = valid ? v : 0.0f;
        float w = (dz ? wz : 1.0f - wz) * (dy ? wy : 1.0f - wy)
                * (dx ? wx : 1.0f - wx);
        out += w * v;
      }
    }
  }
  return out;
}

// ---------------- kernel 1: sampling + row sums ----------------
// grid: (P/256, Q+T, B), block 256
__global__ __launch_bounds__(256)
void sample_kernel(const float* __restrict__ masks,     // [B,Q,64,64,64]
                   const float* __restrict__ tgtvol,    // [B,T,64,64,64]
                   const float* __restrict__ pts,       // [B,P,3]
                   _Float16* __restrict__ predA,        // [B,QPAD,P]
                   _Float16* __restrict__ sigA,         // [B,QPAD,P]
                   _Float16* __restrict__ tgtB,         // [B,TPAD,P]
                   float* __restrict__ sumSp,           // [B,Q]
                   float* __restrict__ sumSg,           // [B,Q]
                   float* __restrict__ sumTg)           // [B,T]
{
  int tid = threadIdx.x;
  int p   = blockIdx.x * 256 + tid;
  int row = blockIdx.y;
  int b   = blockIdx.z;

  size_t pbase = ((size_t)b * P_ + p) * 3;
  float c0 = pts[pbase + 0];
  float c1 = pts[pbase + 1];
  float c2 = pts[pbase + 2];

  float r1v, r2v;
  if (row < Q_) {
    const float* vol = masks + ((size_t)(b * Q_ + row)) * VOL_;
    float v  = sample_tri(vol, c0, c1, c2);
    float sg = 1.0f / (1.0f + expf(-v));
    float sp = fmaxf(v, 0.0f) + log1pf(expf(-fabsf(v)));   // stable softplus
    size_t o = ((size_t)(b * QPAD_ + row)) * P_ + p;
    predA[o] = (_Float16)v;
    sigA[o]  = (_Float16)sg;
    r1v = sp; r2v = sg;
  } else {
    int t = row - Q_;
    const float* vol = tgtvol + ((size_t)(b * T_ + t)) * VOL_;
    float v = sample_tri(vol, c0, c1, c2);
    tgtB[((size_t)(b * TPAD_ + t)) * P_ + p] = (_Float16)v;
    r1v = v; r2v = 0.0f;
  }

  __shared__ float r1[256], r2[256];
  r1[tid] = r1v; r2[tid] = r2v;
  __syncthreads();
  for (int s = 128; s > 0; s >>= 1) {
    if (tid < s) { r1[tid] += r1[tid + s]; r2[tid] += r2[tid + s]; }
    __syncthreads();
  }
  if (tid == 0) {
    if (row < Q_) {
      atomicAdd(&sumSp[b * Q_ + row], r1[0]);
      atomicAdd(&sumSg[b * Q_ + row], r2[0]);
    } else {
      atomicAdd(&sumTg[b * T_ + (row - Q_)], r1[0]);
    }
  }
}

// ---------------- kernel 2: class softmax gather ----------------
// grid: (Q, B), block 128
__global__ __launch_bounds__(128)
void class_cost_kernel(const float* __restrict__ cls,    // [B,Q,L]
                       const int*   __restrict__ labels, // [B,T]
                       float* __restrict__ classProb)    // [B,Q,T]
{
  int q = blockIdx.x, b = blockIdx.y, tid = threadIdx.x;
  const float* lg = cls + ((size_t)(b * Q_ + q)) * L_;
  __shared__ float red[128];

  float mx = -3.402823466e38f;
  for (int i = tid; i < L_; i += 128) mx = fmaxf(mx, lg[i]);
  red[tid] = mx; __syncthreads();
  for (int s = 64; s > 0; s >>= 1) {
    if (tid < s) red[tid] = fmaxf(red[tid], red[tid + s]);
    __syncthreads();
  }
  mx = red[0]; __syncthreads();

  float sum = 0.0f;
  for (int i = tid; i < L_; i += 128) sum += expf(lg[i] - mx);
  red[tid] = sum; __syncthreads();
  for (int s = 64; s > 0; s >>= 1) {
    if (tid < s) red[tid] += red[tid + s];
    __syncthreads();
  }
  float denom = red[0];

  if (tid < T_) {
    int lab = labels[b * T_ + tid];
    if (lab < 0 || lab >= L_) lab = 0;       // defensive
    classProb[((size_t)(b * Q_ + q)) * T_ + tid] = expf(lg[lab] - mx) / denom;
  }
}

// ---------------- kernel 3: dual WMMA GEMM + combine ----------------
// grid: (QPAD/16, TPAD/16, B), block 32 (one wave)
__global__ __launch_bounds__(32)
void gemm_combine(const _Float16* __restrict__ predA,   // [B,QPAD,P]
                  const _Float16* __restrict__ sigA,    // [B,QPAD,P]
                  const _Float16* __restrict__ tgtB,    // [B,TPAD,P]
                  const float* __restrict__ sumSp,
                  const float* __restrict__ sumSg,
                  const float* __restrict__ sumTg,
                  const float* __restrict__ classProb,
                  float* __restrict__ out)              // [B,Q,T]
{
  int mt = blockIdx.x, nt = blockIdx.y, b = blockIdx.z;
  int lane = threadIdx.x;
  int lm = lane & 15;         // row within group
  int hi = lane >> 4;         // lane half-group

  // A (16x32 f16): lanes 0-15 carry K 0-15; lanes 16-31 carry K 16-31
  const _Float16* aPredRow = predA + ((size_t)(b * QPAD_ + mt * 16 + lm)) * P_;
  const _Float16* aSigRow  = sigA  + ((size_t)(b * QPAD_ + mt * 16 + lm)) * P_;
  // B (32x16 f16): N = lane&15; lanes 0-15 carry K 0-15, lanes 16-31 K 16-31
  const _Float16* bRow     = tgtB  + ((size_t)(b * TPAD_ + nt * 16 + lm)) * P_;
  int kA = hi * 16;
  int kB = hi * 16;

  v8f acc1 = {};   // pred . tgt
  v8f acc2 = {};   // sigmoid(pred) . tgt

  for (int k0 = 0; k0 < P_; k0 += 32) {
    const _Float16* ap = aPredRow + k0 + kA;
    const _Float16* sp = aSigRow  + k0 + kA;
    const _Float16* bp = bRow     + k0 + kB;

    __builtin_prefetch(ap + 256, 0, 0);          // global_prefetch_b8
    __builtin_prefetch(bp + 512, 0, 0);

    v8h a0 = *(const v8h*)(ap);
    v8h a1 = *(const v8h*)(ap + 8);
    v8h s0 = *(const v8h*)(sp);
    v8h s1 = *(const v8h*)(sp + 8);
    v16h bb = *(const v16h*)(bp);                // 32B aligned (P%16==0)

    v16h aPred, aSig;
#pragma unroll
    for (int i = 0; i < 8; ++i) {
      aPred[i] = a0[i]; aPred[8 + i] = a1[i];
      aSig[i]  = s0[i]; aSig[8 + i]  = s1[i];
    }

    acc1 = __builtin_amdgcn_wmma_f32_16x16x32_f16(
        false, aPred, false, bb, (short)0, acc1, false, false);
    acc2 = __builtin_amdgcn_wmma_f32_16x16x32_f16(
        false, aSig,  false, bb, (short)0, acc2, false, false);
  }

  // C/D layout: VGPR v, lanes 0-15: M=v, N=lane; lanes 16-31: M=v+8, N=lane-16
  const float invP = 1.0f / (float)P_;
#pragma unroll
  for (int v = 0; v < 8; ++v) {
    int q = mt * 16 + v + hi * 8;
    int t = nt * 16 + lm;
    if (q < Q_ && t < T_) {
      float dotPred = acc1[v];
      float dotSig  = acc2[v];
      float cm = (sumSp[b * Q_ + q] - dotPred) * invP;               // BCE cost
      float cd = 1.0f - (2.0f * dotSig + 1.0f)
                 / (sumSg[b * Q_ + q] + sumTg[b * T_ + t] + 1.0f);   // dice
      float cc = -classProb[((size_t)(b * Q_ + q)) * T_ + t];        // class
      float c = cm + cc + cd;
      if (isnan(c)) c = 0.0f;
      else          c = fminf(fmaxf(c, -CLAMP_), CLAMP_);
      out[((size_t)b * Q_ + q) * T_ + t] = c;
    }
  }
}

// ---------------- host launcher ----------------
extern "C" void kernel_launch(void* const* d_in, const int* in_sizes, int n_in,
                              void* d_out, int out_size, void* d_ws, size_t ws_size,
                              hipStream_t stream) {
  (void)in_sizes; (void)n_in; (void)out_size; (void)ws_size;

  const float* masks  = (const float*)d_in[0];  // [B,Q,64,64,64]
  const float* cls    = (const float*)d_in[1];  // [B,Q,L]
  const float* tgtvol = (const float*)d_in[2];  // [B,T,64,64,64]
  const float* pts    = (const float*)d_in[3];  // [B,P,3]
  const int*   clab   = (const int*)  d_in[4];  // [B,T]
  float* out = (float*)d_out;

  // workspace carve-up (256B aligned)
  char* ws = (char*)d_ws;
  size_t off = 0;
  auto take = [&](size_t bytes) {
    size_t cur = off;
    off = (off + bytes + 255) & ~(size_t)255;
    return cur;
  };
  _Float16* predA    = (_Float16*)(ws + take((size_t)B_ * QPAD_ * P_ * 2));
  _Float16* sigA     = (_Float16*)(ws + take((size_t)B_ * QPAD_ * P_ * 2));
  _Float16* tgtB     = (_Float16*)(ws + take((size_t)B_ * TPAD_ * P_ * 2));
  float*    sumSp    = (float*)(ws + take((size_t)B_ * Q_ * 4));
  float*    sumSg    = (float*)(ws + take((size_t)B_ * Q_ * 4));
  float*    sumTg    = (float*)(ws + take((size_t)B_ * T_ * 4));
  float*    classPrb = (float*)(ws + take((size_t)B_ * Q_ * T_ * 4));
  size_t total = off;

  // zero pad rows + accumulators (capturable async memset)
  hipMemsetAsync(d_ws, 0, total, stream);

  sample_kernel<<<dim3(P_ / 256, Q_ + T_, B_), 256, 0, stream>>>(
      masks, tgtvol, pts, predA, sigA, tgtB, sumSp, sumSg, sumTg);

  class_cost_kernel<<<dim3(Q_, B_), 128, 0, stream>>>(cls, clab, classPrb);

  gemm_combine<<<dim3(QPAD_ / 16, TPAD_ / 16, B_), 32, 0, stream>>>(
      predA, sigA, tgtB, sumSp, sumSg, sumTg, classPrb, out);
}
